// SelfAttLayer_75325136437764
// MI455X (gfx1250) — compile-verified, matrix operand order checked by reading, not run on hardware
//
#include <hip/hip_runtime.h>
#include <hip/hip_bf16.h>

// CDNA5 / gfx1250 fused local self-attention layer (SAN-style).
// B=16, C=64, H=W=64, K=3, heads=8, cph=8.
// Convs run as implicit GEMMs on v_wmma_f32_16x16x32_bf16 (fp32 accum).
// Conv3x3 tile staging uses the Tensor Data Mover (tensor_load_to_lds with
// LDS padding); the fused Q/V 1x1 GEMM uses GLOBAL_LOAD_ASYNC_TO_LDS_B128.

#define AS1 __attribute__((address_space(1)))
#define AS3 __attribute__((address_space(3)))

typedef __attribute__((ext_vector_type(16))) __bf16 bf16x16;
typedef __attribute__((ext_vector_type(8)))  float  fx8;
typedef int      v4i __attribute__((vector_size(16)));   // async builtin param
typedef unsigned vu4 __attribute__((vector_size(16)));   // TDM D# group 0
typedef int      vs4 __attribute__((vector_size(16)));   // TDM D# groups 2/3
typedef int      vs8 __attribute__((vector_size(32)));   // TDM D# group 1 / arg5

union V16 { uint4 u4[2]; bf16x16 v; };

__device__ __forceinline__ unsigned short f2bf(float f) {
    unsigned u = __float_as_uint(f);
    unsigned r = (u + 0x7FFFu + ((u >> 16) & 1u)) >> 16;  // RNE
    return (unsigned short)r;
}
__device__ __forceinline__ unsigned pack2(float a, float b) {
    return (unsigned)f2bf(a) | ((unsigned)f2bf(b) << 16);
}
__device__ __forceinline__ fx8 zf8() {
    fx8 r;
    #pragma unroll
    for (int i = 0; i < 8; ++i) r[i] = 0.0f;
    return r;
}

// Async 16-byte global -> LDS copy (ASYNCcnt tracked, no VGPR data path).
__device__ __forceinline__ void asyncCopy16(const unsigned short* g, unsigned short* l) {
#if __has_builtin(__builtin_amdgcn_global_load_async_to_lds_b128)
    __builtin_amdgcn_global_load_async_to_lds_b128(
        (AS1 v4i*)g, (AS3 v4i*)l, 0, 0);
#else
    unsigned loff = (unsigned)(unsigned long long)(AS3 unsigned short*)l;
    asm volatile("global_load_async_to_lds_b128 %0, %1, off"
                 :: "v"(loff), "v"(g) : "memory");
#endif
}
__device__ __forceinline__ void waitAsync() {
#if __has_builtin(__builtin_amdgcn_s_wait_asynccnt)
    __builtin_amdgcn_s_wait_asynccnt(0);
#else
    asm volatile("s_wait_asynccnt 0x0" ::: "memory");
#endif
}
__device__ __forceinline__ void waitTensor() {
#if __has_builtin(__builtin_amdgcn_s_wait_tensorcnt)
    __builtin_amdgcn_s_wait_tensorcnt(0);
#else
    asm volatile("s_wait_tensorcnt 0x0" ::: "memory");
#endif
}

#if __has_builtin(__builtin_amdgcn_tensor_load_to_lds)
#define HAVE_TDM 1
// TDM copy of a 2-D tile: ncells lines of 64 bf16 (128 B) with
// tensor_dim0_stride = 64 elems, into LDS at 144-B line pitch via the
// descriptor pad feature (pad 16 B after every 128 B stored).
__device__ __forceinline__ void tdmLoadTile(const unsigned short* g,
                                            unsigned ldsByteAddr, int ncells)
{
    unsigned long long ga =
        (unsigned long long)(AS1 const unsigned short*)g;  // global byte addr
    // Group 0: count=1 | lds_addr | global_addr[56:0] | type=2
    vu4 g0 = { 1u,
               ldsByteAddr,
               (unsigned)(ga & 0xFFFFFFFFu),
               (unsigned)(((ga >> 32) & 0x01FFFFFFu) | 0x80000000u) };
    // Group 1: data_size=2B | pad_enable | pad_interval=32DW | pad_amount=4DW,
    // tensor_dim0=64, tensor_dim1=ncells, tile_dim0=64, tile_dim1=ncells,
    // tile_dim2=0, tensor_dim0_stride=64.
    vs8 g1 = { (int)((1u << 16) | (1u << 20) | (4u << 22) | (3u << 25)),
               (int)(64u << 16),                  // tensor_dim0 low16
               (int)((unsigned)ncells << 16),     // dim0 hi=0 | tensor_dim1 lo16
               (int)(64u << 16),                  // dim1 hi=0 | tile_dim0=64
               (int)ncells,                       // tile_dim1 | tile_dim2=0
               64,                                // tensor_dim0_stride low32
               0, 0 };
    vs4 z4 = { 0, 0, 0, 0 };
    vs8 z8 = { 0, 0, 0, 0, 0, 0, 0, 0 };
    __builtin_amdgcn_tensor_load_to_lds(g0, g1, z4, z4, z8, 0);
}
#endif

// ---------------------------------------------------------------------------
// K0a: build zero-padded position-major bf16 tensor
//   xbp[b][pr(66)][pc(72)][c(64)], 1-wide zero halo (cols 66..71 zero pad).
// ---------------------------------------------------------------------------
__global__ __launch_bounds__(256) void convert_kernel(
    const float* __restrict__ x, unsigned short* __restrict__ xbp)
{
    int idx = blockIdx.x * 256 + threadIdx.x;   // 16*66*72*8 = 608256
    int pc = idx % 72;
    int r = idx / 72;
    int pr = r % 66; r /= 66;
    int cg = r & 7;
    int b  = r >> 3;

    int hh = pr - 1, ww = pc - 1;
    bool val = ((unsigned)hh < 64u) && ((unsigned)ww < 64u) && (pc < 66);
    unsigned d[4];
    #pragma unroll
    for (int j = 0; j < 4; ++j) {
        int c0 = cg * 8 + 2 * j;
        float f0 = val ? x[((b * 64 + c0) << 12) + (hh << 6) + ww] : 0.0f;
        float f1 = val ? x[((b * 64 + c0 + 1) << 12) + (hh << 6) + ww] : 0.0f;
        d[j] = pack2(f0, f1);
    }
    uint4* dst = reinterpret_cast<uint4*>(
        xbp + ((size_t)((b * 66 + pr) * 72 + pc) << 6) + cg * 8);
    *dst = make_uint4(d[0], d[1], d[2], d[3]);
}

// ---------------------------------------------------------------------------
// K0b: fold BN into (scale, shift), positional bias, and pre-pack weights
// into per-lane WMMA B-fragment images (frag = 32 lanes x 8 dwords).
// ---------------------------------------------------------------------------
__global__ __launch_bounds__(256) void pack_kernel(
    const float* __restrict__ wq, const float* __restrict__ wk,
    const float* __restrict__ wv,
    const float* __restrict__ pos_h, const float* __restrict__ pos_w,
    const float* __restrict__ qg, const float* __restrict__ qb,
    const float* __restrict__ qm, const float* __restrict__ qv,
    const float* __restrict__ kg, const float* __restrict__ kb,
    const float* __restrict__ km, const float* __restrict__ kv,
    const float* __restrict__ vg, const float* __restrict__ vb,
    const float* __restrict__ vm, const float* __restrict__ vv,
    const float* __restrict__ og, const float* __restrict__ ob,
    const float* __restrict__ om, const float* __restrict__ ov,
    float* __restrict__ coef, float* __restrict__ posb,
    unsigned* __restrict__ wqf, unsigned* __restrict__ wkf,
    unsigned* __restrict__ wvf)
{
    int tid = threadIdx.x;

    if (tid < 256) {
        int s = tid >> 6, n = tid & 63;
        const float* G[4]  = {qg, kg, vg, og};
        const float* Be[4] = {qb, kb, vb, ob};
        const float* M[4]  = {qm, km, vm, om};
        const float* V[4]  = {qv, kv, vv, ov};
        float sc = G[s][n] * rsqrtf(V[s][n] + 1e-5f);
        coef[s * 128 + n] = sc;
        coef[s * 128 + 64 + n] = Be[s][n] - M[s][n] * sc;
    }

    for (int i = tid; i < 576; i += 256) {
        int t = i / 64, c = i & 63;
        int ky = t / 3, kx = t % 3;
        posb[t * 64 + c] = pos_h[c * 3 + ky] + pos_w[c * 3 + kx];
    }

    // 1x1 weights (wq, wv): 8 frags x 256 dwords. W[n][c], K = c.
    for (int i = tid; i < 8 * 256; i += 256) {
        int f = i >> 8, r = i & 255, lane = r >> 3, j = r & 7;
        int kc = f >> 2, nt = f & 3;
        int n = nt * 16 + (lane & 15);
        int klo = (lane >> 4) * 8;
        int kk = (j < 4) ? (klo + 2 * j) : (16 + klo + 2 * (j - 4));
        int c0 = kc * 32 + kk;
        wqf[i] = pack2(wq[n * 64 + c0], wq[n * 64 + c0 + 1]);
        wvf[i] = pack2(wv[n * 64 + c0], wv[n * 64 + c0 + 1]);
    }

    // 3x3 weights: 72 frags (t*2+kc)*4+nt. wk[n][c][ky][kx], K = t*64 + c.
    for (int i = tid; i < 72 * 256; i += 256) {
        int f = i >> 8, r = i & 255, lane = r >> 3, j = r & 7;
        int t = f >> 3, rem = f & 7;
        int kc = rem >> 2, nt = rem & 3;
        int n = nt * 16 + (lane & 15);
        int klo = (lane >> 4) * 8;
        int kk = (j < 4) ? (klo + 2 * j) : (16 + klo + 2 * (j - 4));
        int c0 = kc * 32 + kk;
        wkf[i] = pack2(wk[(n * 64 + c0) * 9 + t], wk[(n * 64 + c0 + 1) * 9 + t]);
    }
}

// ---------------------------------------------------------------------------
// K1: fused Q+V 1x1 conv implicit GEMMs + BN (Q: ReLU). Block = one image row
// (64 pos) x 64 out-ch, 4 waves. One async-staged A tile feeds both weight
// sets (16 WMMAs/wave). LDS cell stride padded to 144 B.
// ---------------------------------------------------------------------------
__global__ __launch_bounds__(128) void gemm1x1_qv_kernel(
    const unsigned short* __restrict__ xbp,
    const unsigned* __restrict__ wqf, const unsigned* __restrict__ wvf,
    const float* __restrict__ coef,   // q at +0, v at +256
    float* __restrict__ qout, float* __restrict__ vout)
{
    __shared__ __align__(16) unsigned short sx[72 * 72];  // 72 cells x 144B

    int tid = threadIdx.x;
    int b = blockIdx.x >> 6, h = blockIdx.x & 63;

    // Stage padded row pr = h+1 (all 72 cols x 64 ch) asynchronously.
    const unsigned short* src = xbp + ((size_t)(b * 66 + h + 1) * 72 << 6);
    for (int i = tid; i < 576; i += 128) {          // 72 cells * 8 chunks
        int cell = i >> 3, j = i & 7;
        asyncCopy16(src + (cell << 6) + j * 8, &sx[cell * 72 + j * 8]);
    }
    waitAsync();
    __syncthreads();

    int wave = tid >> 5, lane = tid & 31;
    int m0 = wave * 16;
    int mA = m0 + (lane & 15);
    int klo = (lane >> 4) * 8;

    fx8 accq[4], accv[4];
    #pragma unroll
    for (int nt = 0; nt < 4; ++nt) { accq[nt] = zf8(); accv[nt] = zf8(); }

    #pragma unroll
    for (int kc = 0; kc < 2; ++kc) {
        const uint4* ap = reinterpret_cast<const uint4*>(
            &sx[(mA + 1) * 72 + kc * 32 + klo]);   // pc = m+1 (halo col 0)
        V16 a;
        a.u4[0] = ap[0];   // K klo..klo+7
        a.u4[1] = ap[2];   // K klo+16..klo+23
        #pragma unroll
        for (int nt = 0; nt < 4; ++nt) {
            int f = kc * 4 + nt;
            const uint4* bq = reinterpret_cast<const uint4*>(wqf) + (f * 32 + lane) * 2;
            const uint4* bv = reinterpret_cast<const uint4*>(wvf) + (f * 32 + lane) * 2;
            V16 wb;
            wb.u4[0] = bq[0]; wb.u4[1] = bq[1];
            accq[nt] = __builtin_amdgcn_wmma_f32_16x16x32_bf16(
                false, a.v, false, wb.v, (short)0, accq[nt], false, false);
            wb.u4[0] = bv[0]; wb.u4[1] = bv[1];
            accv[nt] = __builtin_amdgcn_wmma_f32_16x16x32_bf16(
                false, a.v, false, wb.v, (short)0, accv[nt], false, false);
        }
    }

    int ms = m0 + ((lane >> 4) << 3);
    #pragma unroll
    for (int nt = 0; nt < 4; ++nt) {
        int n = nt * 16 + (lane & 15);
        float sq = coef[n],       tq = coef[64 + n];
        float sv = coef[256 + n], tv = coef[320 + n];
        float vq[8], vv_[8];
        #pragma unroll
        for (int r = 0; r < 8; ++r) {
            vq[r]  = fmaxf(accq[nt][r] * sq + tq, 0.0f);   // ReLU on Q
            vv_[r] = accv[nt][r] * sv + tv;                // no ReLU on V
        }
        float* oq = qout + ((b * 64 + n) << 12) + (h << 6) + ms;
        float* ov = vout + ((b * 64 + n) << 12) + (h << 6) + ms;
        *reinterpret_cast<float4*>(oq)     = make_float4(vq[0], vq[1], vq[2], vq[3]);
        *reinterpret_cast<float4*>(oq + 4) = make_float4(vq[4], vq[5], vq[6], vq[7]);
        *reinterpret_cast<float4*>(ov)     = make_float4(vv_[0], vv_[1], vv_[2], vv_[3]);
        *reinterpret_cast<float4*>(ov + 4) = make_float4(vv_[4], vv_[5], vv_[6], vv_[7]);
    }
}

// ---------------------------------------------------------------------------
// K2: 3x3 conv (pad=1) implicit GEMM (K = 9 taps x 64 ch) + BN + ReLU.
// Block = one image row. The 3 contiguous padded rows (216 lines x 128 B) are
// staged by ONE tensor_load_to_lds (TDM) with LDS line padding to 144 B;
// 72 WMMAs per wave.
// ---------------------------------------------------------------------------
__global__ __launch_bounds__(128) void conv3x3_kernel(
    const unsigned short* __restrict__ xbp, const unsigned* __restrict__ wkf,
    const float* __restrict__ coef, float* __restrict__ out)
{
    __shared__ __align__(16) unsigned short sx[216 * 72];  // 216 cells x 144B

    int tid = threadIdx.x;
    int b = blockIdx.x >> 6, h = blockIdx.x & 63;

    const unsigned short* src = xbp + ((size_t)(b * 66 + h) * 72 << 6);
#ifdef HAVE_TDM
    if (tid < 32) {   // one wave issues the DMA (TDM ignores EXEC)
        unsigned ldsAddr = (unsigned)(unsigned long long)(AS3 unsigned short*)&sx[0];
        tdmLoadTile(src, ldsAddr, 216);
    }
    waitTensor();
#else
    for (int i = tid; i < 1728; i += 128) {
        int cell = i >> 3, j = i & 7;
        asyncCopy16(src + (cell << 6) + j * 8, &sx[cell * 72 + j * 8]);
    }
    waitAsync();
#endif
    __syncthreads();

    int wave = tid >> 5, lane = tid & 31;
    int m0 = wave * 16;
    int mA = m0 + (lane & 15);
    int klo = (lane >> 4) * 8;

    fx8 acc[4];
    #pragma unroll
    for (int nt = 0; nt < 4; ++nt) acc[nt] = zf8();

    #pragma unroll
    for (int t = 0; t < 9; ++t) {
        int ky = t / 3, kx = t % 3;
        int cell = ky * 72 + mA + kx;   // pc = m + kx (halo col 0)
        #pragma unroll
        for (int kc = 0; kc < 2; ++kc) {
            const uint4* ap = reinterpret_cast<const uint4*>(
                &sx[cell * 72 + kc * 32 + klo]);
            V16 a;
            a.u4[0] = ap[0];
            a.u4[1] = ap[2];
            #pragma unroll
            for (int nt = 0; nt < 4; ++nt) {
                int f = (t * 2 + kc) * 4 + nt;
                const uint4* bp = reinterpret_cast<const uint4*>(wkf) + (f * 32 + lane) * 2;
                V16 bv;
                bv.u4[0] = bp[0];
                bv.u4[1] = bp[1];
                acc[nt] = __builtin_amdgcn_wmma_f32_16x16x32_bf16(
                    false, a.v, false, bv.v, (short)0, acc[nt], false, false);
            }
        }
    }

    int ms = m0 + ((lane >> 4) << 3);
    #pragma unroll
    for (int nt = 0; nt < 4; ++nt) {
        int n = nt * 16 + (lane & 15);
        float sc = coef[n], sh = coef[64 + n];
        float vals[8];
        #pragma unroll
        for (int r = 0; r < 8; ++r)
            vals[r] = fmaxf(acc[nt][r] * sc + sh, 0.0f);   // ReLU
        float* op = out + ((b * 64 + n) << 12) + (h << 6) + ms;
        *reinterpret_cast<float4*>(op)     = make_float4(vals[0], vals[1], vals[2], vals[3]);
        *reinterpret_cast<float4*>(op + 4) = make_float4(vals[4], vals[5], vals[6], vals[7]);
    }
}

// ---------------------------------------------------------------------------
// K3: attention logits (pos bias added after zero-pad unfold), softmax over
// 9 taps, V aggregation, output BN + ReLU. One thread per (b, head, position).
// ---------------------------------------------------------------------------
__global__ __launch_bounds__(256) void attn_kernel(
    const float* __restrict__ q, const float* __restrict__ k,
    const float* __restrict__ v, const float* __restrict__ posb,
    const float* __restrict__ ocoef, float* __restrict__ out)
{
    int g = blockIdx.x * 256 + threadIdx.x;   // 0 .. 16*8*4096-1
    int p = g & 4095;
    int bh = g >> 12;
    int hd = bh & 7, b = bh >> 3;
    int h = p >> 6, w = p & 63;
    int cbase = b * 64 + hd * 8;

    float qv[8];
    const float* qp = q + (cbase << 12) + p;
    #pragma unroll
    for (int c = 0; c < 8; ++c) qv[c] = qp[c << 12];

    float logit[9];
    #pragma unroll
    for (int t = 0; t < 9; ++t) {
        int ky = t / 3, kx = t % 3;
        int h2 = h + ky - 1, w2 = w + kx - 1;
        bool valid = ((unsigned)h2 < 64u) && ((unsigned)w2 < 64u);
        const float* kp = k + (cbase << 12) + (h2 << 6) + w2;
        const float* pp = posb + t * 64 + hd * 8;
        float s = 0.0f;
        #pragma unroll
        for (int c = 0; c < 8; ++c) {
            float kk = valid ? kp[c << 12] : 0.0f;
            s += qv[c] * (kk + pp[c]);
        }
        logit[t] = s;
    }

    float mx = logit[0];
    #pragma unroll
    for (int t = 1; t < 9; ++t) mx = fmaxf(mx, logit[t]);
    float e[9], se = 0.0f;
    #pragma unroll
    for (int t = 0; t < 9; ++t) { e[t] = __expf(logit[t] - mx); se += e[t]; }
    float inv = 1.0f / se;

    float o[8];
    #pragma unroll
    for (int c = 0; c < 8; ++c) o[c] = 0.0f;
    #pragma unroll
    for (int t = 0; t < 9; ++t) {
        int ky = t / 3, kx = t % 3;
        int h2 = h + ky - 1, w2 = w + kx - 1;
        if (((unsigned)h2 < 64u) && ((unsigned)w2 < 64u)) {
            const float* vp = v + (cbase << 12) + (h2 << 6) + w2;
            float a = e[t] * inv;
            #pragma unroll
            for (int c = 0; c < 8; ++c) o[c] += a * vp[c << 12];
        }
    }

    #pragma unroll
    for (int c = 0; c < 8; ++c) {
        int n = hd * 8 + c;
        float y = o[c] * ocoef[n] + ocoef[64 + n];
        out[((b * 64 + n) << 12) + p] = fmaxf(y, 0.0f);
    }
}

// ---------------------------------------------------------------------------
extern "C" void kernel_launch(void* const* d_in, const int* in_sizes, int n_in,
                              void* d_out, int out_size, void* d_ws, size_t ws_size,
                              hipStream_t stream)
{
    const float* x     = (const float*)d_in[0];
    const float* wq    = (const float*)d_in[1];
    const float* wk    = (const float*)d_in[2];
    const float* wv    = (const float*)d_in[3];
    const float* pos_h = (const float*)d_in[4];
    const float* pos_w = (const float*)d_in[5];
    const float* qg = (const float*)d_in[6],  *qb = (const float*)d_in[7];
    const float* qm = (const float*)d_in[8],  *qv = (const float*)d_in[9];
    const float* kg = (const float*)d_in[10], *kb = (const float*)d_in[11];
    const float* km = (const float*)d_in[12], *kv = (const float*)d_in[13];
    const float* vg = (const float*)d_in[14], *vb = (const float*)d_in[15];
    const float* vm = (const float*)d_in[16], *vv = (const float*)d_in[17];
    const float* og = (const float*)d_in[18], *ob = (const float*)d_in[19];
    const float* om = (const float*)d_in[20], *ov = (const float*)d_in[21];

    char* ws = (char*)d_ws;
    float*    coef = (float*)(ws);                 //  2048 B (4 sets x 128 f)
    float*    posb = (float*)(ws + 2048);          //  2304 B
    unsigned* wqf  = (unsigned*)(ws + 4608);       //  8192 B
    unsigned* wvf  = (unsigned*)(ws + 12800);      //  8192 B
    unsigned* wkf  = (unsigned*)(ws + 20992);      // 73728 B
    unsigned short* xbp = (unsigned short*)(ws + 98304);  // 9,732,096 B
    const size_t PLANE = 16777216;                 // 16*64*4096*4 bytes
    float* qbuf = (float*)(ws + 9830400);
    float* vbuf = (float*)(ws + 9830400 + PLANE);
    float* kbuf = (float*)(ws + 9830400 + 2 * PLANE);

    convert_kernel<<<2376, 256, 0, stream>>>(x, xbp);

    pack_kernel<<<1, 256, 0, stream>>>(wq, wk, wv, pos_h, pos_w,
                                       qg, qb, qm, qv, kg, kb, km, kv,
                                       vg, vb, vm, vv, og, ob, om, ov,
                                       coef, posb, wqf, wkf, wvf);

    gemm1x1_qv_kernel<<<1024, 128, 0, stream>>>(xbp, wqf, wvf, coef, qbuf, vbuf);
    conv3x3_kernel<<<1024, 128, 0, stream>>>(xbp, wkf, coef + 128, kbuf);

    attn_kernel<<<2048, 256, 0, stream>>>(qbuf, kbuf, vbuf, posb,
                                          coef + 384, (float*)d_out);
}
